// EmergentSpinGlass_40553081209136
// MI455X (gfx1250) — matrix operand order repaired and checked by reading, not compiled, not run on hardware
//
#include <hip/hip_runtime.h>
#include <hip/hip_bf16.h>
#include <math.h>

typedef __attribute__((ext_vector_type(16))) __bf16 v16bf;
typedef __attribute__((ext_vector_type(8)))  __bf16 v8bf;
typedef __attribute__((ext_vector_type(8)))  float  v8f;

#define MODE_BIAS 1
#define MODE_F32  2
#define MODE_BF16 4
#define MODE_EPI  8

// ---------------------------------------------------------------------------
// WMMA fragment loaders for v_wmma_f32_16x16x32_bf16 (wave32).
// A (16x32, MxK): lanes 0-15 row M=lane, elems 0..7 = K k0..k0+7, elems 8..15
//   = K k0+16..k0+23; lanes 16-31 same rows, K ranges shifted by +8.
// B (32x16, KxN): lanes 0-15 col N=lane, elems 0..7 = K k0..k0+7, elems 8..15
//   = K k0+16..k0+23; lanes 16-31 same cols, K ranges shifted by +8 (matches A).
// Both A and "B" source matrices are row-major with K contiguous (NT GEMM).
// ---------------------------------------------------------------------------
__device__ __forceinline__ v16bf load_a_frag(const __bf16* __restrict__ A, int lda,
                                             int row0, int k0, int lane) {
  int r  = row0 + (lane & 15);
  int kc = k0 + ((lane >> 4) << 3);
  const __bf16* p = A + (size_t)r * lda + kc;
  v8bf lo = *(const v8bf*)p;
  v8bf hi = *(const v8bf*)(p + 16);
  v16bf f;
#pragma unroll
  for (int i = 0; i < 8; ++i) { f[i] = lo[i]; f[i + 8] = hi[i]; }
  return f;
}

__device__ __forceinline__ v16bf load_b_frag(const __bf16* __restrict__ Bw, int ldb,
                                             int col0, int k0, int lane) {
  int c  = col0 + (lane & 15);
  int kc = k0 + ((lane >> 4) << 3);
  const __bf16* p = Bw + (size_t)c * ldb + kc;
  v8bf lo = *(const v8bf*)p;
  v8bf hi = *(const v8bf*)(p + 16);
  v16bf f;
#pragma unroll
  for (int i = 0; i < 8; ++i) { f[i] = lo[i]; f[i + 8] = hi[i]; }
  return f;
}

// ---------------------------------------------------------------------------
// Generic NT GEMM:  C[m,n] = alpha * sum_k A[m,k]*Bw[n,k]  (+ bias[n])
// Block = 256 threads (8 waves: 2 along M x 4 along N), block tile 128x256,
// per-wave tile 64x64 (4x4 wmma accumulators = 128 VGPRs).
// Per k-step (32): 8 fragment loads (16 b128) feed 16 WMMAs -> 32 FLOP/byte.
// mode: bias add, f32 store, bf16 store, fused tanh epilogue.
// Batched via blockIdx.z with element strides.
// ---------------------------------------------------------------------------
__global__ __launch_bounds__(256)
void gemm_nt_wmma(const __bf16* __restrict__ A, const __bf16* __restrict__ Bw,
                  const float* __restrict__ bias,
                  float* __restrict__ Cf, __bf16* __restrict__ Cb,
                  const float* __restrict__ sAdd, const float* __restrict__ giAdd,
                  const float* __restrict__ beta, float* __restrict__ outF,
                  int M, int Nn, int K, int lda, int ldb, int ldc,
                  long long strideA, long long strideB, long long strideC,
                  long long strideS, long long strideGi,
                  float alpha, int mode)
{
  (void)M; (void)Nn;
  const int bz   = blockIdx.z;
  const __bf16* Ab = A  + (size_t)bz * strideA;
  const __bf16* Bb = Bw + (size_t)bz * strideB;

  const int lane = threadIdx.x & 31;
  const int wid  = threadIdx.x >> 5;
  const int wm   = wid & 1;       // 2 waves along M
  const int wn   = wid >> 1;      // 4 waves along N
  const int row0 = blockIdx.y * 128 + wm * 64;
  const int col0 = blockIdx.x * 256 + wn * 64;

  v8f acc[4][4];
#pragma unroll
  for (int i = 0; i < 4; ++i)
#pragma unroll
    for (int j = 0; j < 4; ++j) acc[i][j] = (v8f)(0.0f);

  for (int k = 0; k < K; k += 32) {
    v16bf a0 = load_a_frag(Ab, lda, row0,      k, lane);
    v16bf a1 = load_a_frag(Ab, lda, row0 + 16, k, lane);
    v16bf a2 = load_a_frag(Ab, lda, row0 + 32, k, lane);
    v16bf a3 = load_a_frag(Ab, lda, row0 + 48, k, lane);
    v16bf b0 = load_b_frag(Bb, ldb, col0,      k, lane);
    v16bf b1 = load_b_frag(Bb, ldb, col0 + 16, k, lane);
    v16bf b2 = load_b_frag(Bb, ldb, col0 + 32, k, lane);
    v16bf b3 = load_b_frag(Bb, ldb, col0 + 48, k, lane);
    acc[0][0] = __builtin_amdgcn_wmma_f32_16x16x32_bf16(false, a0, false, b0, (short)0, acc[0][0], false, false);
    acc[0][1] = __builtin_amdgcn_wmma_f32_16x16x32_bf16(false, a0, false, b1, (short)0, acc[0][1], false, false);
    acc[0][2] = __builtin_amdgcn_wmma_f32_16x16x32_bf16(false, a0, false, b2, (short)0, acc[0][2], false, false);
    acc[0][3] = __builtin_amdgcn_wmma_f32_16x16x32_bf16(false, a0, false, b3, (short)0, acc[0][3], false, false);
    acc[1][0] = __builtin_amdgcn_wmma_f32_16x16x32_bf16(false, a1, false, b0, (short)0, acc[1][0], false, false);
    acc[1][1] = __builtin_amdgcn_wmma_f32_16x16x32_bf16(false, a1, false, b1, (short)0, acc[1][1], false, false);
    acc[1][2] = __builtin_amdgcn_wmma_f32_16x16x32_bf16(false, a1, false, b2, (short)0, acc[1][2], false, false);
    acc[1][3] = __builtin_amdgcn_wmma_f32_16x16x32_bf16(false, a1, false, b3, (short)0, acc[1][3], false, false);
    acc[2][0] = __builtin_amdgcn_wmma_f32_16x16x32_bf16(false, a2, false, b0, (short)0, acc[2][0], false, false);
    acc[2][1] = __builtin_amdgcn_wmma_f32_16x16x32_bf16(false, a2, false, b1, (short)0, acc[2][1], false, false);
    acc[2][2] = __builtin_amdgcn_wmma_f32_16x16x32_bf16(false, a2, false, b2, (short)0, acc[2][2], false, false);
    acc[2][3] = __builtin_amdgcn_wmma_f32_16x16x32_bf16(false, a2, false, b3, (short)0, acc[2][3], false, false);
    acc[3][0] = __builtin_amdgcn_wmma_f32_16x16x32_bf16(false, a3, false, b0, (short)0, acc[3][0], false, false);
    acc[3][1] = __builtin_amdgcn_wmma_f32_16x16x32_bf16(false, a3, false, b1, (short)0, acc[3][1], false, false);
    acc[3][2] = __builtin_amdgcn_wmma_f32_16x16x32_bf16(false, a3, false, b2, (short)0, acc[3][2], false, false);
    acc[3][3] = __builtin_amdgcn_wmma_f32_16x16x32_bf16(false, a3, false, b3, (short)0, acc[3][3], false, false);
  }

  // C/D layout: VGPR r, lanes 0-15 -> row r, col lane; lanes 16-31 -> row r+8.
  const int rbase = (lane >> 4) << 3;
  const int cloc  = lane & 15;
#pragma unroll
  for (int i = 0; i < 4; ++i) {
#pragma unroll
    for (int j = 0; j < 4; ++j) {
      const int col = col0 + j * 16 + cloc;
      const float bv = (mode & MODE_BIAS) ? bias[col] : 0.0f;
#pragma unroll
      for (int r = 0; r < 8; ++r) {
        const int row = row0 + i * 16 + rbase + r;
        const float v = acc[i][j][r] * alpha + bv;
        const size_t idx = (size_t)row * ldc + col;
        if (mode & MODE_F32)  Cf[(size_t)bz * strideC + idx] = v;
        if (mode & MODE_BF16) Cb[(size_t)bz * strideC + idx] = (__bf16)v;
        if (mode & MODE_EPI) {
          const float sv = sAdd[(size_t)bz * strideS + idx];
          const float gv = giAdd[(size_t)bz * strideGi + col];
          outF[(size_t)bz * strideC + idx] = tanhf(beta[0] * (sv + gv + v));
        }
      }
    }
  }
}

// ---------------------------------------------------------------------------
// Helpers
// ---------------------------------------------------------------------------
__global__ void f32_to_bf16_kernel(const float* __restrict__ in, __bf16* __restrict__ o, size_t n) {
  size_t i = (size_t)blockIdx.x * blockDim.x + threadIdx.x;
  const size_t stride = (size_t)gridDim.x * blockDim.x;
  for (; i < n; i += stride) o[i] = (__bf16)in[i];
}

__global__ void mean_kernel(const float* __restrict__ s, float* __restrict__ meanb,
                            int Nn, int Dd) {
  const int idx = blockIdx.x * blockDim.x + threadIdx.x;   // over B*D
  const int b = idx / Dd, d = idx - b * Dd;
  const float* p = s + (size_t)b * Nn * Dd + d;
  float acc = 0.f;
  for (int n = 0; n < Nn; ++n) acc += p[(size_t)n * Dd];
  meanb[idx] = acc / (float)Nn;
}

__global__ void global_influence_kernel(const float* __restrict__ meanb,
                                        const float* __restrict__ Wg,
                                        float* __restrict__ gib, int Dd) {
  const int idx = blockIdx.x * blockDim.x + threadIdx.x;   // over B*D
  const int b = idx / Dd, o = idx - b * Dd;
  const float* m    = meanb + (size_t)b * Dd;
  const float* wrow = Wg    + (size_t)o * Dd;
  float acc = 0.f;
  for (int d = 0; d < Dd; ++d) acc += m[d] * wrow[d];
  gib[idx] = acc;
}

__global__ void transpose_bf16(const __bf16* __restrict__ in, __bf16* __restrict__ out,
                               int rows, int cols) {  // in: rows x cols -> out: cols x rows
  __shared__ __bf16 tile[32][33];
  const int b = blockIdx.z;
  in  += (size_t)b * rows * cols;
  out += (size_t)b * rows * cols;
  const int c0 = blockIdx.x * 32, r0 = blockIdx.y * 32;
  for (int i = threadIdx.y; i < 32; i += 8)
    tile[i][threadIdx.x] = in[(size_t)(r0 + i) * cols + c0 + threadIdx.x];
  __syncthreads();
  for (int i = threadIdx.y; i < 32; i += 8)
    out[(size_t)(c0 + i) * rows + r0 + threadIdx.x] = tile[threadIdx.x][i];
}

__global__ __launch_bounds__(256)
void softmax_rows(const float* __restrict__ scores, __bf16* __restrict__ aff, int Ncols) {
  __shared__ float red[256];
  const size_t row = blockIdx.x;
  const float* p = scores + row * Ncols;   // Ncols == 2048 == 8*256
  const int t = threadIdx.x;
  float vals[8];
  float m = -INFINITY;
#pragma unroll
  for (int i = 0; i < 8; ++i) { vals[i] = p[t + i * 256]; m = fmaxf(m, vals[i]); }
  red[t] = m; __syncthreads();
  for (int s = 128; s > 0; s >>= 1) { if (t < s) red[t] = fmaxf(red[t], red[t + s]); __syncthreads(); }
  m = red[0]; __syncthreads();
  float sum = 0.f;
#pragma unroll
  for (int i = 0; i < 8; ++i) { vals[i] = __expf(vals[i] - m); sum += vals[i]; }
  red[t] = sum; __syncthreads();
  for (int s = 128; s > 0; s >>= 1) { if (t < s) red[t] += red[t + s]; __syncthreads(); }
  const float inv = 1.0f / red[0];
  __bf16* q = aff + row * Ncols;
#pragma unroll
  for (int i = 0; i < 8; ++i) q[t + i * 256] = (__bf16)(vals[i] * inv);
}

// ---------------------------------------------------------------------------
extern "C" void kernel_launch(void* const* d_in, const int* in_sizes, int n_in,
                              void* d_out, int out_size, void* d_ws, size_t ws_size,
                              hipStream_t stream) {
  (void)in_sizes; (void)n_in; (void)out_size; (void)ws_size;
  const float* x        = (const float*)d_in[0];
  const float* W_spin   = (const float*)d_in[1];
  const float* b_spin   = (const float*)d_in[2];
  const float* W_global = (const float*)d_in[3];
  const float* W_J      = (const float*)d_in[4];
  const float* beta     = (const float*)d_in[5];
  float* out = (float*)d_out;

  const int Bb = 4, Nn = 2048, Dd = 1024;
  const size_t BND = (size_t)Bb * Nn * Dd;   // 8M elems
  const size_t BNN = (size_t)Bb * Nn * Nn;   // 16M elems
  const size_t DD  = (size_t)Dd * Dd;

  char* w = (char*)d_ws;
  auto carve = [&](size_t bytes) { char* p = w; w += (bytes + 255) & ~(size_t)255; return p; };
  __bf16* x_bf   = (__bf16*)carve(BND * 2);
  __bf16* Ws_bf  = (__bf16*)carve(DD * 2);
  __bf16* WJ_bf  = (__bf16*)carve(DD * 2);
  float*  s_f32  = (float*) carve(BND * 4);
  __bf16* s_bf   = (__bf16*)carve(BND * 2);
  __bf16* sT_bf  = (__bf16*)carve(BND * 2);
  __bf16* e_bf   = (__bf16*)carve(BND * 2);
  float*  scores = (float*) carve(BNN * 4);
  __bf16* aff    = (__bf16*)carve(BNN * 2);
  float*  meanb  = (float*) carve((size_t)Bb * Dd * 4);
  float*  gib    = (float*) carve((size_t)Bb * Dd * 4);

  // 1) precision down-converts for WMMA operands
  f32_to_bf16_kernel<<<2048, 256, 0, stream>>>(x, x_bf, BND);
  f32_to_bf16_kernel<<<512, 256, 0, stream>>>(W_spin, Ws_bf, DD);
  f32_to_bf16_kernel<<<512, 256, 0, stream>>>(W_J, WJ_bf, DD);

  // 2) s = x @ W_spin^T + b   (M=8192, N=1024, K=1024), keep f32 + bf16 copies
  gemm_nt_wmma<<<dim3(Dd / 256, (Bb * Nn) / 128, 1), 256, 0, stream>>>(
      x_bf, Ws_bf, b_spin, s_f32, s_bf, nullptr, nullptr, nullptr, nullptr,
      Bb * Nn, Dd, Dd, Dd, Dd, Dd, 0, 0, 0, 0, 0, 1.0f,
      MODE_BIAS | MODE_F32 | MODE_BF16);

  // 3) mean field + global influence (tiny)
  mean_kernel<<<(Bb * Dd) / 256, 256, 0, stream>>>(s_f32, meanb, Nn, Dd);
  global_influence_kernel<<<(Bb * Dd) / 256, 256, 0, stream>>>(meanb, W_global, gib, Dd);

  // 4) e = (s @ W_J^T) * 1/sqrt(D)  -> bf16 (scale folded at f32 accumulate)
  gemm_nt_wmma<<<dim3(Dd / 256, (Bb * Nn) / 128, 1), 256, 0, stream>>>(
      s_bf, WJ_bf, nullptr, nullptr, e_bf, nullptr, nullptr, nullptr, nullptr,
      Bb * Nn, Dd, Dd, Dd, Dd, Dd, 0, 0, 0, 0, 0, 0.03125f, MODE_BF16);

  // 5) s^T per batch for the PV matmul (NT form needs K contiguous)
  transpose_bf16<<<dim3(Dd / 32, Nn / 32, Bb), dim3(32, 8), 0, stream>>>(s_bf, sT_bf, Nn, Dd);

  // 6) scores[b] = e[b] @ s[b]^T   (batched; M=N=2048, K=1024)
  gemm_nt_wmma<<<dim3(Nn / 256, Nn / 128, Bb), 256, 0, stream>>>(
      e_bf, s_bf, nullptr, scores, nullptr, nullptr, nullptr, nullptr, nullptr,
      Nn, Nn, Dd, Dd, Dd, Nn,
      (long long)Nn * Dd, (long long)Nn * Dd, (long long)Nn * Nn, 0, 0,
      1.0f, MODE_F32);

  // 7) row softmax -> bf16 affinities
  softmax_rows<<<Bb * Nn, 256, 0, stream>>>(scores, aff, Nn);

  // 8) out = tanh(beta*(s + gi + aff @ s))   (batched; M=2048, N=1024, K=2048)
  gemm_nt_wmma<<<dim3(Dd / 256, Nn / 128, Bb), 256, 0, stream>>>(
      aff, sT_bf, nullptr, nullptr, nullptr, s_f32, gib, beta, out,
      Nn, Dd, Nn, Nn, Nn, Dd,
      (long long)Nn * Nn, (long long)Nn * Dd, (long long)Nn * Dd,
      (long long)Nn * Dd, (long long)Dd,
      1.0f, MODE_EPI);
}
